// FullModel_50663434224461
// MI455X (gfx1250) — matrix operand order repaired and checked by reading, not compiled, not run on hardware
//
#include <hip/hip_runtime.h>
#include <hip/hip_bf16.h>

// ---------------------------------------------------------------------------
// Problem constants (from reference)
// ---------------------------------------------------------------------------
constexpr int B_    = 256;
constexpr int NCLS_ = 1000;
constexpr int P_    = 5;
constexpr int C_    = 72;
constexpr int D_    = 512;
constexpr int H_    = 8;
constexpr int DIMG_ = 768;
constexpr int T_    = P_ + C_;   // 77
constexpr int DH_   = D_ / H_;   // 64

// ---------------------------------------------------------------------------
// WMMA vector types (gfx1250, wave32)
// ---------------------------------------------------------------------------
typedef __attribute__((ext_vector_type(16))) _Float16 v16h;
typedef __attribute__((ext_vector_type(8)))  _Float16 v8h;
typedef __attribute__((ext_vector_type(8)))  float    v8f;

// A-source modes for the fused-LN GEMM
enum { AM_PLAIN = 0, AM_LN1 = 1, AM_LN1_LAST = 2, AM_LN1_SHARED = 3, AM_LN2 = 4 };

// ---------------------------------------------------------------------------
// CDNA5 async copy helpers (ASYNCcnt-tracked global->LDS DMA, no VGPR bounce)
// LDS byte address = low 32 bits of the generic pointer (ISA 10.2 aperture).
// ---------------------------------------------------------------------------
__device__ __forceinline__ void async_copy_b128(void* lds_dst, const void* gsrc) {
    unsigned lds_addr = (unsigned)(uintptr_t)lds_dst;
    unsigned long long gaddr = (unsigned long long)(uintptr_t)gsrc;
    asm volatile("global_load_async_to_lds_b128 %0, %1, off"
                 :
                 : "v"(lds_addr), "v"(gaddr)
                 : "memory");
}
__device__ __forceinline__ void wait_asynccnt0() {
    asm volatile("s_wait_asynccnt 0" ::: "memory");
}

// ---------------------------------------------------------------------------
// Weight convert + transpose: W[K,N] f32 -> Wt[N,K] f16 (done once; B tiles
// then stream with contiguous-K rows that match the WMMA B fragment layout)
// ---------------------------------------------------------------------------
__global__ void convert_wt_kernel(const float* __restrict__ W,
                                  _Float16* __restrict__ Wt, int K, int N) {
    int idx = blockIdx.x * blockDim.x + threadIdx.x;
    if (idx >= N * K) return;
    int n = idx / K;
    int k = idx - n * K;
    Wt[idx] = (_Float16)W[(size_t)k * N + n];
}

// ---------------------------------------------------------------------------
// LayerNorm statistics, pass 1: one wave32 per prompt token row
// ---------------------------------------------------------------------------
__global__ __launch_bounds__(256) void ln_stats1_kernel(
    const float* __restrict__ ctx, const float* __restrict__ cls_tok,
    float* __restrict__ mean, float* __restrict__ rstd) {
    int row  = blockIdx.x * 8 + (threadIdx.x >> 5);
    int lane = threadIdx.x & 31;
    if (row >= NCLS_ * T_) return;
    int cls = row / T_;
    int tt  = row - cls * T_;
    const float* x = (tt < P_) ? ctx + ((size_t)cls * P_ + tt) * D_
                               : cls_tok + ((size_t)cls * C_ + (tt - P_)) * D_;
    float s = 0.f, s2 = 0.f;
    for (int k = lane; k < D_; k += 32) { float v = x[k]; s += v; s2 += v * v; }
    for (int o = 16; o; o >>= 1) { s += __shfl_xor(s, o, 32); s2 += __shfl_xor(s2, o, 32); }
    if (lane == 0) {
        float m   = s * (1.0f / D_);
        float var = s2 * (1.0f / D_) - m * m;
        mean[row] = m;
        rstd[row] = rsqrtf(var + 1e-5f);
    }
}

// LayerNorm statistics, pass 2: only the P ctx tokens per class, scaled by attr
__global__ __launch_bounds__(256) void ln_stats2_kernel(
    const float* __restrict__ ctx, const float* __restrict__ attr,
    float* __restrict__ mean, float* __restrict__ rstd) {
    int row  = blockIdx.x * 8 + (threadIdx.x >> 5);
    int lane = threadIdx.x & 31;
    if (row >= NCLS_ * P_) return;
    const float* x = ctx + (size_t)row * D_;
    float a = attr[row];
    float s = 0.f, s2 = 0.f;
    for (int k = lane; k < D_; k += 32) { float v = x[k] * a; s += v; s2 += v * v; }
    for (int o = 16; o; o >>= 1) { s += __shfl_xor(s, o, 32); s2 += __shfl_xor(s2, o, 32); }
    if (lane == 0) {
        float m   = s * (1.0f / D_);
        float var = s2 * (1.0f / D_) - m * m;
        mean[row] = m;
        rstd[row] = rsqrtf(var + 1e-5f);
    }
}

// ---------------------------------------------------------------------------
// Fused-LN WMMA GEMM: C[M,N] = A'[M,K] @ Bt[N,K]^T
//   A' rows produced on the fly (plain f32 / LN variants) -> f16 in LDS.
//   Bt is pre-transposed f16 [N][K]; its tiles stream via async global->LDS.
// Tile 128x128x64, 256 threads = 8 waves, wave tile 64x32 = 4x2 WMMA frags.
// Double-buffered LDS, one barrier per k-step.
// ---------------------------------------------------------------------------
#define BM  128
#define BN  128
#define BK  64
#define LDK 72   // padded LDS stride in halves: 16B-aligned, conflict-free

template <int AMODE>
__device__ __forceinline__ void a_row_info(
    int r, const float* Ap, int K,
    const float* ctx, const float* cls_tok,
    const float* mean, const float* rstd, const float* attr,
    const float*& src, float& mu, float& rs, float& sc) {
    if constexpr (AMODE == AM_PLAIN) {
        src = Ap + (size_t)r * K; mu = 0.f; rs = 1.f; sc = 1.f;
    } else if constexpr (AMODE == AM_LN1) {
        int cls = r / T_; int tt = r - cls * T_;
        src = (tt < P_) ? ctx + ((size_t)cls * P_ + tt) * D_
                        : cls_tok + ((size_t)cls * C_ + (tt - P_)) * D_;
        mu = mean[r]; rs = rstd[r]; sc = 1.f;
    } else if constexpr (AMODE == AM_LN1_LAST) {
        int idx = r * T_ + (T_ - 1);
        src = cls_tok + ((size_t)r * C_ + (C_ - 1)) * D_;
        mu = mean[idx]; rs = rstd[idx]; sc = 1.f;
    } else if constexpr (AMODE == AM_LN1_SHARED) {
        int cls = r / C_; int tt = r - cls * C_;          // tt in [0,C)
        int idx = cls * T_ + P_ + tt;
        src = cls_tok + ((size_t)cls * C_ + tt) * D_;
        mu = mean[idx]; rs = rstd[idx]; sc = 1.f;
    } else { // AM_LN2: r over NCLS*P, ctx rows indexed directly
        src = ctx + (size_t)r * D_;
        mu = mean[r]; rs = rstd[r]; sc = attr[r];
    }
}

template <int AMODE, bool OUT_F16>
__global__ __launch_bounds__(256) void gemm_wmma_kernel(
    const float* __restrict__ Ap,
    const float* __restrict__ ctx, const float* __restrict__ cls_tok,
    const float* __restrict__ lng, const float* __restrict__ lnb,
    const float* __restrict__ mean, const float* __restrict__ rstd,
    const float* __restrict__ attr,
    const _Float16* __restrict__ Bt,
    float* __restrict__ Cf, _Float16* __restrict__ Ch,
    const float* __restrict__ addPtr, size_t addStride,
    const float* __restrict__ scalePtr,
    int M, int N, int K) {
    __shared__ _Float16 As[2][BM * LDK];
    __shared__ _Float16 Bs[2][BN * LDK];

    const int tid  = threadIdx.x;
    const int wave = tid >> 5;
    const int lane = tid & 31;
    const int la   = lane & 15;
    const int lh   = lane >> 4;
    const int wm   = wave >> 2;   // 0..1
    const int wn   = wave & 3;    // 0..3
    const int m0   = blockIdx.y * BM;
    const int n0   = blockIdx.x * BN;

    // --- per-thread tile-load assignments (2 threads per row, 32 cols each)
    const int ar    = tid >> 1;
    const int aseg  = tid & 1;
    const int rGlob = m0 + ar;
    const bool arow_ok = rGlob < M;
    const float* src = nullptr; float mu = 0.f, rs = 1.f, sc = 1.f;
    if (arow_ok)
        a_row_info<AMODE>(rGlob, Ap, K, ctx, cls_tok, mean, rstd, attr, src, mu, rs, sc);

    const int nGlob = n0 + ar;
    const bool brow_ok = nGlob < N;
    const _Float16* bsrc = Bt + (size_t)nGlob * K;

    float4 a_pre[8];  // prefetched raw f32 A row chunk (32 floats)

    // --- helpers -----------------------------------------------------------
    auto load_A_regs = [&](int k0) {
        if (arow_ok) {
            const int kb = k0 + aseg * 32;
#pragma unroll
            for (int i = 0; i < 8; ++i)
                a_pre[i] = *(const float4*)(src + kb + i * 4);
        }
    };
    auto store_A_tile = [&](int buf, int k0) {
        _Float16* dst = &As[buf][ar * LDK + aseg * 32];
        const int kb = k0 + aseg * 32;
        if (arow_ok) {
#pragma unroll
            for (int i = 0; i < 8; ++i) {
                float vals[4] = {a_pre[i].x, a_pre[i].y, a_pre[i].z, a_pre[i].w};
                if constexpr (AMODE != AM_PLAIN) {
                    float4 gv = *(const float4*)(lng + kb + i * 4);
                    float4 bv = *(const float4*)(lnb + kb + i * 4);
                    vals[0] = (vals[0] * sc - mu) * rs * gv.x + bv.x;
                    vals[1] = (vals[1] * sc - mu) * rs * gv.y + bv.y;
                    vals[2] = (vals[2] * sc - mu) * rs * gv.z + bv.z;
                    vals[3] = (vals[3] * sc - mu) * rs * gv.w + bv.w;
                }
                dst[i * 4 + 0] = (_Float16)vals[0];
                dst[i * 4 + 1] = (_Float16)vals[1];
                dst[i * 4 + 2] = (_Float16)vals[2];
                dst[i * 4 + 3] = (_Float16)vals[3];
            }
        } else {
#pragma unroll
            for (int i = 0; i < 32; ++i) dst[i] = (_Float16)0.f;
        }
    };
    auto issue_B_tile = [&](int buf, int k0) {
        _Float16* dst = &Bs[buf][ar * LDK + aseg * 32];
        if (brow_ok) {
            const _Float16* g = bsrc + k0 + aseg * 32;   // 64B contiguous
#pragma unroll
            for (int i = 0; i < 4; ++i)
                async_copy_b128(dst + i * 8, g + i * 8); // ASYNCcnt-tracked DMA
        } else {
#pragma unroll
            for (int i = 0; i < 32; ++i) dst[i] = (_Float16)0.f;
        }
    };

    // --- accumulators ------------------------------------------------------
    v8f acc[4][2];
#pragma unroll
    for (int i = 0; i < 4; ++i)
#pragma unroll
        for (int j = 0; j < 2; ++j)
            acc[i][j] = (v8f){0.f, 0.f, 0.f, 0.f, 0.f, 0.f, 0.f, 0.f};

    const int nk = K / BK;

    // --- prologue: fill buffer 0 ------------------------------------------
    issue_B_tile(0, 0);
    load_A_regs(0);
    store_A_tile(0, 0);
    wait_asynccnt0();
    __syncthreads();

    // --- main pipeline: prefetch j+1 while computing j --------------------
    for (int j = 0; j < nk; ++j) {
        const int rbuf = j & 1;
        const int wbuf = rbuf ^ 1;
        if (j + 1 < nk) {
            const int k0n = (j + 1) * BK;
            issue_B_tile(wbuf, k0n);   // async DMA overlaps the WMMAs below
            load_A_regs(k0n);
            store_A_tile(wbuf, k0n);
        }

#pragma unroll
        for (int ks = 0; ks < 2; ++ks) {
            // A fragments (ISA 7.12.2): lane row = lane&15, elems 0..7 = K
            // kb..kb+7, elems 8..15 = kb+16..kb+23 with kb = (lane>>4)*8.
            v16h af[4];
#pragma unroll
            for (int mi = 0; mi < 4; ++mi) {
                const _Float16* p =
                    &As[rbuf][(wm * 64 + mi * 16 + la) * LDK + ks * 32 + lh * 8];
                v8h lo = *(const v8h*)p;
                v8h hi = *(const v8h*)(p + 16);
                v16h f;
#pragma unroll
                for (int e = 0; e < 8; ++e) { f[e] = lo[e]; f[e + 8] = hi[e]; }
                af[mi] = f;
            }
            // B fragments: lane column = lane&15, same K order as A:
            // elems 0..7 = kb..kb+7, elems 8..15 = kb+16..kb+23, kb = (lane>>4)*8.
            v16h bf[2];
#pragma unroll
            for (int ni = 0; ni < 2; ++ni) {
                const _Float16* p =
                    &Bs[rbuf][(wn * 32 + ni * 16 + la) * LDK + ks * 32 + lh * 8];
                v8h lo = *(const v8h*)p;
                v8h hi = *(const v8h*)(p + 16);
                v16h f;
#pragma unroll
                for (int e = 0; e < 8; ++e) { f[e] = lo[e]; f[e + 8] = hi[e]; }
                bf[ni] = f;
            }
#pragma unroll
            for (int mi = 0; mi < 4; ++mi)
#pragma unroll
                for (int ni = 0; ni < 2; ++ni)
                    acc[mi][ni] = __builtin_amdgcn_wmma_f32_16x16x32_f16(
                        false, af[mi], false, bf[ni], (short)0, acc[mi][ni], false, false);
        }

        wait_asynccnt0();   // own DMA into wbuf has landed
        __syncthreads();    // everyone done reading rbuf / writing wbuf
    }

    // --- epilogue ----------------------------------------------------------
    const float scl = scalePtr ? __expf(scalePtr[0]) : 1.0f;
#pragma unroll
    for (int mi = 0; mi < 4; ++mi) {
#pragma unroll
        for (int ni = 0; ni < 2; ++ni) {
            const int bn = n0 + wn * 32 + ni * 16 + la;
            if (bn >= N) continue;
            const int bm = m0 + wm * 64 + mi * 16 + lh * 8;
#pragma unroll
            for (int r = 0; r < 8; ++r) {
                int row = bm + r;
                if (row >= M) break;
                float v = acc[mi][ni][r] * scl;
                if (addPtr) v += addPtr[(size_t)row * addStride + bn];
                if (OUT_F16) Ch[(size_t)row * N + bn] = (_Float16)v;
                else         Cf[(size_t)row * N + bn] = v;
            }
        }
    }
}

// ---------------------------------------------------------------------------
// Pass-1 attention: per class, scores(last-q, all k) per head, softmax,
// head-mean over first P keys, softmax over P -> attr[NCLS,P]
// ---------------------------------------------------------------------------
__global__ __launch_bounds__(256) void attn1_kernel(
    const float* __restrict__ qlast, const _Float16* __restrict__ kall,
    float* __restrict__ attr) {
    int cls = blockIdx.x;
    __shared__ float sc[H_][T_];
    __shared__ float aP[H_][P_];
    const float* q = qlast + (size_t)cls * D_;
    const _Float16* kb = kall + (size_t)cls * T_ * D_;
    for (int i = threadIdx.x; i < T_ * H_; i += 256) {
        int t = i / H_;
        int h = i - t * H_;
        const float* qp = q + h * DH_;
        const _Float16* kp = kb + (size_t)t * D_ + h * DH_;
        float s = 0.f;
        for (int d = 0; d < DH_; ++d) s += qp[d] * (float)kp[d];
        sc[h][t] = s * 0.125f;   // 1/sqrt(DH)
    }
    __syncthreads();
    int h = threadIdx.x >> 5, lane = threadIdx.x & 31;
    float mx = -1e30f;
    for (int t = lane; t < T_; t += 32) mx = fmaxf(mx, sc[h][t]);
    for (int o = 16; o; o >>= 1) mx = fmaxf(mx, __shfl_xor(mx, o, 32));
    float sum = 0.f;
    for (int t = lane; t < T_; t += 32) sum += __expf(sc[h][t] - mx);
    for (int o = 16; o; o >>= 1) sum += __shfl_xor(sum, o, 32);
    if (lane < P_) aP[h][lane] = __expf(sc[h][lane] - mx) / sum;
    __syncthreads();
    if (threadIdx.x == 0) {
        float vv[P_]; float m2 = -1e30f;
        for (int t = 0; t < P_; ++t) {
            float s = 0.f;
            for (int hh = 0; hh < H_; ++hh) s += aP[hh][t];
            vv[t] = s * (1.0f / H_);
            m2 = fmaxf(m2, vv[t]);
        }
        float ss = 0.f;
        for (int t = 0; t < P_; ++t) { vv[t] = __expf(vv[t] - m2); ss += vv[t]; }
        for (int t = 0; t < P_; ++t) attr[(size_t)cls * P_ + t] = vv[t] / ss;
    }
}

// ---------------------------------------------------------------------------
// Pass-2 attention + output: keys/values for t<P come from the re-LN'd ctx,
// t>=P reuse pass-1 k / shared v. Emits o_last[NCLS, D].
// ---------------------------------------------------------------------------
__global__ __launch_bounds__(256) void attn2_kernel(
    const float* __restrict__ qlast, const _Float16* __restrict__ kall,
    const _Float16* __restrict__ kctx2, const _Float16* __restrict__ vctx2,
    const _Float16* __restrict__ vshr, float* __restrict__ olast) {
    int cls = blockIdx.x;
    __shared__ float sc[H_][T_];
    const float* q = qlast + (size_t)cls * D_;
    for (int i = threadIdx.x; i < T_ * H_; i += 256) {
        int t = i / H_;
        int h = i - t * H_;
        const float* qp = q + h * DH_;
        const _Float16* kp = (t < P_)
            ? kctx2 + ((size_t)cls * P_ + t) * D_ + h * DH_
            : kall  + ((size_t)cls * T_ + t) * D_ + h * DH_;
        float s = 0.f;
        for (int d = 0; d < DH_; ++d) s += qp[d] * (float)kp[d];
        sc[h][t] = s * 0.125f;
    }
    __syncthreads();
    int h = threadIdx.x >> 5, lane = threadIdx.x & 31;
    float mx = -1e30f;
    for (int t = lane; t < T_; t += 32) mx = fmaxf(mx, sc[h][t]);
    for (int o = 16; o; o >>= 1) mx = fmaxf(mx, __shfl_xor(mx, o, 32));
    float sum = 0.f;
    for (int t = lane; t < T_; t += 32) { float e = __expf(sc[h][t] - mx); sc[h][t] = e; sum += e; }
    for (int o = 16; o; o >>= 1) sum += __shfl_xor(sum, o, 32);
    float inv = 1.0f / sum;
    for (int t = lane; t < T_; t += 32) sc[h][t] *= inv;
    __syncthreads();
#pragma unroll
    for (int j = 0; j < 2; ++j) {
        int d  = threadIdx.x * 2 + j;      // 0..511
        int hd = d >> 6;
        float acc = 0.f;
        for (int t = 0; t < T_; ++t) {
            float a = sc[hd][t];
            float v = (t < P_) ? (float)vctx2[((size_t)cls * P_ + t) * D_ + d]
                               : (float)vshr[((size_t)cls * C_ + (t - P_)) * D_ + d];
            acc += a * v;
        }
        olast[(size_t)cls * D_ + d] = acc;
    }
}

// ---------------------------------------------------------------------------
// Row L2 normalize (D=512). Optionally also emits an f16 [rows][D] copy,
// which doubles as the pre-transposed B matrix for the logits GEMM.
// ---------------------------------------------------------------------------
__global__ __launch_bounds__(256) void normalize_rows_kernel(
    const float* __restrict__ in, float* __restrict__ outf,
    _Float16* __restrict__ outh, int rows) {
    int row  = blockIdx.x * 8 + (threadIdx.x >> 5);
    int lane = threadIdx.x & 31;
    if (row >= rows) return;
    const float* x = in + (size_t)row * D_;
    float s = 0.f;
    for (int k = lane; k < D_; k += 32) { float v = x[k]; s += v * v; }
    for (int o = 16; o; o >>= 1) s += __shfl_xor(s, o, 32);
    float inv = rsqrtf(s);
    for (int k = lane; k < D_; k += 32) {
        float v = x[k] * inv;
        outf[(size_t)row * D_ + k] = v;
        if (outh) outh[(size_t)row * D_ + k] = (_Float16)v;
    }
}

// ---------------------------------------------------------------------------
// Host-side orchestration
// ---------------------------------------------------------------------------
static inline int ceil_div(int a, int b) { return (a + b - 1) / b; }

extern "C" void kernel_launch(void* const* d_in, const int* in_sizes, int n_in,
                              void* d_out, int out_size, void* d_ws, size_t ws_size,
                              hipStream_t stream) {
    (void)in_sizes; (void)n_in; (void)out_size; (void)ws_size;
    const float* images    = (const float*)d_in[0];   // [B, DIMG]
    const float* W_img     = (const float*)d_in[1];   // [DIMG, D]
    const float* ctx       = (const float*)d_in[2];   // [NCLS, P, D]
    const float* cls_tok   = (const float*)d_in[3];   // [NCLS, C, D]
    const float* ln_g      = (const float*)d_in[4];   // [D]
    const float* ln_b      = (const float*)d_in[5];   // [D]
    const float* Wq        = (const float*)d_in[6];   // [D, D]
    const float* Wk        = (const float*)d_in[7];
    const float* Wv        = (const float*)d_in[8];
    const float* Wo        = (const float*)d_in[9];
    const float* text_proj = (const float*)d_in[10];
    const float* logit_sc  = (const float*)d_in[11];  // [1]
    float* logits = (float*)d_out;                    // [B, NCLS]

    // bump allocator over workspace (256B aligned slots)
    char* wsp = (char*)d_ws;
    auto alloc = [&](size_t bytes) -> void* {
        void* p = wsp;
        wsp += (bytes + 255) & ~(size_t)255;
        return p;
    };
    _Float16* wimg_h = (_Float16*)alloc((size_t)D_ * DIMG_ * 2);   // [D][DIMG]
    _Float16* wq_h   = (_Float16*)alloc((size_t)D_ * D_ * 2);      // [N][K]
    _Float16* wk_h   = (_Float16*)alloc((size_t)D_ * D_ * 2);
    _Float16* wv_h   = (_Float16*)alloc((size_t)D_ * D_ * 2);
    _Float16* wo_h   = (_Float16*)alloc((size_t)D_ * D_ * 2);
    _Float16* wp_h   = (_Float16*)alloc((size_t)D_ * D_ * 2);
    float* mean1 = (float*)alloc((size_t)NCLS_ * T_ * 4);
    float* rstd1 = (float*)alloc((size_t)NCLS_ * T_ * 4);
    float* mean2 = (float*)alloc((size_t)NCLS_ * P_ * 4);
    float* rstd2 = (float*)alloc((size_t)NCLS_ * P_ * 4);
    _Float16* kall  = (_Float16*)alloc((size_t)NCLS_ * T_ * D_ * 2);
    _Float16* vshr  = (_Float16*)alloc((size_t)NCLS_ * C_ * D_ * 2);
    _Float16* kctx2 = (_Float16*)alloc((size_t)NCLS_ * P_ * D_ * 2);
    _Float16* vctx2 = (_Float16*)alloc((size_t)NCLS_ * P_ * D_ * 2);
    float* qlast  = (float*)alloc((size_t)NCLS_ * D_ * 4);
    float* attr   = (float*)alloc((size_t)NCLS_ * P_ * 4);
    float* olast  = (float*)alloc((size_t)NCLS_ * D_ * 4);
    float* hlast  = (float*)alloc((size_t)NCLS_ * D_ * 4);
    float* txtraw = (float*)alloc((size_t)NCLS_ * D_ * 4);
    float* txt    = (float*)alloc((size_t)NCLS_ * D_ * 4);
    _Float16* txth = (_Float16*)alloc((size_t)NCLS_ * D_ * 2);
    float* imgraw = (float*)alloc((size_t)B_ * D_ * 4);
    float* img    = (float*)alloc((size_t)B_ * D_ * 4);

    // ---- 1. weight conversion (once per call; amortized over ~5100 GEMM tiles)
    convert_wt_kernel<<<ceil_div(D_ * DIMG_, 256), 256, 0, stream>>>(W_img, wimg_h, DIMG_, D_);
    convert_wt_kernel<<<ceil_div(D_ * D_, 256), 256, 0, stream>>>(Wq, wq_h, D_, D_);
    convert_wt_kernel<<<ceil_div(D_ * D_, 256), 256, 0, stream>>>(Wk, wk_h, D_, D_);
    convert_wt_kernel<<<ceil_div(D_ * D_, 256), 256, 0, stream>>>(Wv, wv_h, D_, D_);
    convert_wt_kernel<<<ceil_div(D_ * D_, 256), 256, 0, stream>>>(Wo, wo_h, D_, D_);
    convert_wt_kernel<<<ceil_div(D_ * D_, 256), 256, 0, stream>>>(text_proj, wp_h, D_, D_);

    // ---- 2. LN stats pass 1 (all 77k prompt tokens)
    ln_stats1_kernel<<<ceil_div(NCLS_ * T_, 8), 256, 0, stream>>>(ctx, cls_tok, mean1, rstd1);

    // ---- 3. k_all = LN(prompt) @ Wk   [77000,512] f16
    gemm_wmma_kernel<AM_LN1, true><<<dim3(ceil_div(D_, BN), ceil_div(NCLS_ * T_, BM)), 256, 0, stream>>>(
        nullptr, ctx, cls_tok, ln_g, ln_b, mean1, rstd1, nullptr,
        wk_h, nullptr, kall, nullptr, 0, nullptr, NCLS_ * T_, D_, D_);

    // ---- 4. q_last = LN(prompt)[:, -1] @ Wq  (shared by both passes)
    gemm_wmma_kernel<AM_LN1_LAST, false><<<dim3(ceil_div(D_, BN), ceil_div(NCLS_, BM)), 256, 0, stream>>>(
        nullptr, ctx, cls_tok, ln_g, ln_b, mean1, rstd1, nullptr,
        wq_h, qlast, nullptr, nullptr, 0, nullptr, NCLS_, D_, D_);

    // ---- 5. pass-1 attention -> attribution weights
    attn1_kernel<<<NCLS_, 256, 0, stream>>>(qlast, kall, attr);

    // ---- 6. LN stats pass 2 (attr-scaled ctx tokens only)
    ln_stats2_kernel<<<ceil_div(NCLS_ * P_, 8), 256, 0, stream>>>(ctx, attr, mean2, rstd2);

    // ---- 7. v_shared = LN(prompt)[:, P:] @ Wv  (identical across passes)
    gemm_wmma_kernel<AM_LN1_SHARED, true><<<dim3(ceil_div(D_, BN), ceil_div(NCLS_ * C_, BM)), 256, 0, stream>>>(
        nullptr, ctx, cls_tok, ln_g, ln_b, mean1, rstd1, nullptr,
        wv_h, nullptr, vshr, nullptr, 0, nullptr, NCLS_ * C_, D_, D_);

    // ---- 8. pass-2 k,v for the P adjusted ctx tokens
    gemm_wmma_kernel<AM_LN2, true><<<dim3(ceil_div(D_, BN), ceil_div(NCLS_ * P_, BM)), 256, 0, stream>>>(
        nullptr, ctx, cls_tok, ln_g, ln_b, mean2, rstd2, attr,
        wk_h, nullptr, kctx2, nullptr, 0, nullptr, NCLS_ * P_, D_, D_);
    gemm_wmma_kernel<AM_LN2, true><<<dim3(ceil_div(D_, BN), ceil_div(NCLS_ * P_, BM)), 256, 0, stream>>>(
        nullptr, ctx, cls_tok, ln_g, ln_b, mean2, rstd2, attr,
        wv_h, nullptr, vctx2, nullptr, 0, nullptr, NCLS_ * P_, D_, D_);

    // ---- 9. pass-2 attention, last-token output
    attn2_kernel<<<NCLS_, 256, 0, stream>>>(qlast, kall, kctx2, vctx2, vshr, olast);

    // ---- 10. h_last = cls_tok[:, -1] + o_last @ Wo  (residual fused)
    gemm_wmma_kernel<AM_PLAIN, false><<<dim3(ceil_div(D_, BN), ceil_div(NCLS_, BM)), 256, 0, stream>>>(
        olast, nullptr, nullptr, nullptr, nullptr, nullptr, nullptr, nullptr,
        wo_h, hlast, nullptr,
        cls_tok + (size_t)(C_ - 1) * D_, (size_t)C_ * D_,
        nullptr, NCLS_, D_, D_);

    // ---- 11. txt = normalize(h_last @ text_proj); f16 copy is logits-B
    gemm_wmma_kernel<AM_PLAIN, false><<<dim3(ceil_div(D_, BN), ceil_div(NCLS_, BM)), 256, 0, stream>>>(
        hlast, nullptr, nullptr, nullptr, nullptr, nullptr, nullptr, nullptr,
        wp_h, txtraw, nullptr, nullptr, 0, nullptr, NCLS_, D_, D_);
    normalize_rows_kernel<<<ceil_div(NCLS_, 8), 256, 0, stream>>>(txtraw, txt, txth, NCLS_);

    // ---- 12. image branch: img = normalize(images @ W_img)
    gemm_wmma_kernel<AM_PLAIN, false><<<dim3(ceil_div(D_, BN), ceil_div(B_, BM)), 256, 0, stream>>>(
        images, nullptr, nullptr, nullptr, nullptr, nullptr, nullptr, nullptr,
        wimg_h, imgraw, nullptr, nullptr, 0, nullptr, B_, D_, DIMG_);
    normalize_rows_kernel<<<ceil_div(B_, 8), 256, 0, stream>>>(imgraw, img, nullptr, B_);

    // ---- 13. logits = exp(ls) * img @ txt^T   (txth is [NCLS][D] = B [N][K])
    gemm_wmma_kernel<AM_PLAIN, false><<<dim3(ceil_div(NCLS_, BN), ceil_div(B_, BM)), 256, 0, stream>>>(
        img, nullptr, nullptr, nullptr, nullptr, nullptr, nullptr, nullptr,
        txth, logits, nullptr, nullptr, 0, logit_sc, B_, NCLS_, D_);
}
